// Decoder_31344671326374
// MI455X (gfx1250) — compile-verified, matrix operand order checked by reading, not run on hardware
//
#include <hip/hip_runtime.h>
#include <hip/hip_bf16.h>

// ---------------------------------------------------------------------------
// Problem constants (reference: B=64, S=128, T=32, E=H=512, V=32000)
// ---------------------------------------------------------------------------
#define Bq   64
#define Sq   128
#define Tq   32
#define Hq   512
#define Eq   512
#define Vq   32000
#define TWOH 1024
#define KIH  1536   // E + 2H  (rnn_in width, also 3H)
#define KOUT 2048   // H + 2H + E (out_feat width)

typedef __attribute__((ext_vector_type(16))) __bf16 v16bf;
typedef __attribute__((ext_vector_type(8)))  __bf16 v8bf;
typedef __attribute__((ext_vector_type(8)))  float  v8f;

// ---------------------------------------------------------------------------
// Generic bf16 WMMA GEMM:  C[M,N] = A[M,K] * B[N,K]^T (+ bias[N])
// 256 threads = 8 waves (2 wave-rows x 4 wave-cols); wave tile = 64x32 C,
// i.e. 4x2 accumulators of v_wmma_f32_16x16x32_bf16.
// Requires: N % 128 == 0, K % 32 == 0, M % 64 == 0, lda/ldb % 8 == 0.
// ---------------------------------------------------------------------------
__global__ __launch_bounds__(256) void gemm_bf16_tn(
    const __bf16* __restrict__ A, int lda,
    const __bf16* __restrict__ Bm, int ldb,
    float* __restrict__ C, int ldc,
    int M, int N, int K, const float* __restrict__ bias)
{
  const int lane = threadIdx.x & 31;
  const int w    = threadIdx.x >> 5;   // 0..7
  const int wr   = w >> 2;             // 0..1
  const int wc   = w & 3;              // 0..3
  const int m0   = blockIdx.y * 128 + wr * 64;
  const int n0   = blockIdx.x * 128 + wc * 32;
  if (m0 >= M) return;                 // M is a multiple of 64 for all callers

  v8f acc[4][2] = {};

  const int rsel = lane & 15;          // row within 16-row tile
  const int koff = (lane >> 4) << 3;   // lanes 16-31 handle K+8 / K+24

  for (int k = 0; k < K; k += 32) {
    v16bf a[4], b[2];
#pragma unroll
    for (int i = 0; i < 4; ++i) {
      const __bf16* p = A + (size_t)(m0 + 16 * i + rsel) * lda + k + koff;
      v16bf f;
      ((v8bf*)&f)[0] = *(const v8bf*)p;         // K+0..7   -> VGPRs 0..3
      ((v8bf*)&f)[1] = *(const v8bf*)(p + 16);  // K+16..23 -> VGPRs 4..7
      a[i] = f;
    }
#pragma unroll
    for (int j = 0; j < 2; ++j) {
      const __bf16* p = Bm + (size_t)(n0 + 16 * j + rsel) * ldb + k + koff;
      v16bf f;
      ((v8bf*)&f)[0] = *(const v8bf*)p;
      ((v8bf*)&f)[1] = *(const v8bf*)(p + 16);
      b[j] = f;
    }
#pragma unroll
    for (int i = 0; i < 4; ++i)
#pragma unroll
      for (int j = 0; j < 2; ++j)
        acc[i][j] = __builtin_amdgcn_wmma_f32_16x16x32_bf16(
            false, a[i], false, b[j], (short)0, acc[i][j], false, false);
  }

  // C/D layout: VGPR e -> row (e + 8*(lane>=16)), col = lane&15 within tile
  const int cn   = lane & 15;
  const int radd = (lane >> 4) << 3;
#pragma unroll
  for (int i = 0; i < 4; ++i) {
#pragma unroll
    for (int j = 0; j < 2; ++j) {
      const int col = n0 + 16 * j + cn;
      const float bv = bias ? bias[col] : 0.0f;
#pragma unroll
      for (int e = 0; e < 8; ++e) {
        const int row = m0 + 16 * i + radd + e;
        C[(size_t)row * ldc + col] = acc[i][j][e] + bv;
      }
    }
  }
}

// ---------------------------------------------------------------------------
// Elementwise helpers
// ---------------------------------------------------------------------------
__global__ void cvt_f32_bf16(const float* __restrict__ s, __bf16* __restrict__ d, long n) {
  long i = (long)blockIdx.x * blockDim.x + threadIdx.x;
  long st = (long)gridDim.x * blockDim.x;
  for (; i < n; i += st) d[i] = (__bf16)s[i];
}

// encoder_out [S,B,2H] -> enc_bt_bf [B,S,2H] (bf16)
__global__ void enc_transpose_bf16(const float* __restrict__ enc, __bf16* __restrict__ d) {
  const long n = (long)Bq * Sq * TWOH;
  long i = (long)blockIdx.x * blockDim.x + threadIdx.x;
  long st = (long)gridDim.x * blockDim.x;
  for (; i < n; i += st) {
    int dd = (int)(i & (TWOH - 1));
    int s  = (int)((i >> 10) & (Sq - 1));
    int b  = (int)(i >> 17);
    d[i] = (__bf16)enc[((size_t)s * Bq + b) * TWOH + dd];
  }
}

__global__ void zero_f32(float* __restrict__ p, long n) {
  long i = (long)blockIdx.x * blockDim.x + threadIdx.x;
  long st = (long)gridDim.x * blockDim.x;
  for (; i < n; i += st) p[i] = 0.0f;
}

// x = emb[trg[:,t]] (PAD row -> 0); write bf16 into rnn_in[:,0:E] and out_feat[:,1536:2048]
__global__ void gather_x(const int* __restrict__ trg, const float* __restrict__ emb,
                         int t, __bf16* __restrict__ rnnin_bf, __bf16* __restrict__ outfeat_row) {
  const int b   = blockIdx.x;
  const int tok = trg[b * Tq + t];
  for (int e = threadIdx.x; e < Eq; e += blockDim.x) {
    float v = (tok == 0) ? 0.0f : emb[(size_t)tok * Eq + e];
    __bf16 xb = (__bf16)v;
    rnnin_bf[(size_t)b * KIH + e] = xb;
    outfeat_row[(size_t)b * KOUT + (Hq + TWOH) + e] = xb;
  }
}

// Fused attention: scores -> masked softmax -> context. One block per batch b.
// energy = tanh(h_attn[b,:] + enc_proj[b,s,:]) ; score = energy . v
// enc_proj already contains attn_b (folded in as GEMM bias).
__global__ __launch_bounds__(Sq) void attn_fused(
    const float* __restrict__ hattn, const float* __restrict__ encproj,
    const float* __restrict__ vvec, const int* __restrict__ src_tokens,
    const float* __restrict__ encoder_out, float* __restrict__ weighted,
    __bf16* __restrict__ rnnin_bf, __bf16* __restrict__ outfeat_row)
{
  const int b = blockIdx.x;
  const int s = threadIdx.x;            // 0..127, one score per thread
  __shared__ float red[Sq];
  __shared__ float sattn[Sq];

  const float* hp = hattn + (size_t)b * Hq;
  const float* ep = encproj + ((size_t)b * Sq + s) * Hq;
  float sc = 0.0f;
  for (int h = 0; h < Hq; ++h) sc += tanhf(hp[h] + ep[h]) * vvec[h];
  const bool valid = (src_tokens[b * Sq + s] != 0);
  sc = valid ? sc : -1e30f;

  red[s] = sc; __syncthreads();
  for (int st = Sq / 2; st > 0; st >>= 1) {
    if (s < st) red[s] = fmaxf(red[s], red[s + st]);
    __syncthreads();
  }
  const float mx = red[0]; __syncthreads();

  const float ex = valid ? __expf(sc - mx) : 0.0f;
  red[s] = ex; __syncthreads();
  for (int st = Sq / 2; st > 0; st >>= 1) {
    if (s < st) red[s] += red[s + st];
    __syncthreads();
  }
  sattn[s] = ex / red[0];
  __syncthreads();

  // weighted[b,d] = sum_s attn[s] * encoder_out[s,b,d]
  for (int d = s; d < TWOH; d += Sq) {
    float acc = 0.0f;
    for (int ss = 0; ss < Sq; ++ss)
      acc += sattn[ss] * encoder_out[((size_t)ss * Bq + b) * TWOH + d];
    weighted[(size_t)b * TWOH + d] = acc;
    __bf16 wb = (__bf16)acc;
    rnnin_bf[(size_t)b * KIH + Eq + d] = wb;
    outfeat_row[(size_t)b * KOUT + Hq + d] = wb;
  }
}

// GRU gate fusion (torch gate order r,z,n); updates h in place, emits bf16 h_new
__global__ void gru_update(const float* __restrict__ gi, const float* __restrict__ gh,
                           float* __restrict__ h, __bf16* __restrict__ outfeat_row) {
  const long n = (long)Bq * Hq;
  long i = (long)blockIdx.x * blockDim.x + threadIdx.x;
  long st = (long)gridDim.x * blockDim.x;
  for (; i < n; i += st) {
    const int b = (int)(i >> 9);
    const int j = (int)(i & (Hq - 1));
    const size_t gb = (size_t)b * KIH;
    float ir = gi[gb + j],            hr = gh[gb + j];
    float iz = gi[gb + Hq + j],       hz = gh[gb + Hq + j];
    float in = gi[gb + 2 * Hq + j],   hn = gh[gb + 2 * Hq + j];
    float r = 1.0f / (1.0f + __expf(-(ir + hr)));
    float z = 1.0f / (1.0f + __expf(-(iz + hz)));
    float nn = tanhf(in + r * hn);
    float hv = h[i];
    float hnew = (1.0f - z) * nn + z * hv;
    h[i] = hnew;
    outfeat_row[(size_t)b * KOUT + j] = (__bf16)hnew;
  }
}

// ---------------------------------------------------------------------------
// Host side
// ---------------------------------------------------------------------------
static inline int gblocks(long n) {
  long b = (n + 255) / 256;
  return (int)(b > 8192 ? 8192 : (b < 1 ? 1 : b));
}

extern "C" void kernel_launch(void* const* d_in, const int* in_sizes, int n_in,
                              void* d_out, int out_size, void* d_ws, size_t ws_size,
                              hipStream_t stream) {
  (void)in_sizes; (void)n_in; (void)out_size; (void)ws_size;

  const int*   trg    = (const int*)  d_in[0];
  const int*   src    = (const int*)  d_in[1];
  const float* enc    = (const float*)d_in[2];   // [S,B,2H]
  const float* hidden = (const float*)d_in[3];   // [B,H]
  const float* embed  = (const float*)d_in[4];   // [V,E]
  const float* attnW  = (const float*)d_in[5];   // [H, 3H]
  const float* attnb  = (const float*)d_in[6];   // [H]
  const float* vvec   = (const float*)d_in[7];   // [H]
  const float* Wih    = (const float*)d_in[8];   // [3H, 1536]
  const float* Whh    = (const float*)d_in[9];   // [3H, 512]
  const float* bih    = (const float*)d_in[10];
  const float* bhh    = (const float*)d_in[11];
  const float* outW   = (const float*)d_in[12];  // [V, 2048]
  const float* outb   = (const float*)d_in[13];
  float* out = (float*)d_out;                    // [T,B,V]

  // ---- workspace layout (256B aligned slices) ----
  char* ws = (char*)d_ws;
  size_t off = 0;
  auto take = [&](size_t bytes) -> char* {
    char* p = ws + off;
    off = (off + bytes + 255) & ~(size_t)255;
    return p;
  };
  __bf16* outW_bf    = (__bf16*)take((size_t)Vq * KOUT * 2);        // 131 MB (L2-resident)
  __bf16* Wih_bf     = (__bf16*)take((size_t)KIH * KIH * 2);
  __bf16* Whh_bf     = (__bf16*)take((size_t)KIH * Hq * 2);
  __bf16* attnW_bf   = (__bf16*)take((size_t)Hq * KIH * 2);
  __bf16* encbt_bf   = (__bf16*)take((size_t)Bq * Sq * TWOH * 2);
  float*  encproj    = (float*) take((size_t)Bq * Sq * Hq * 4);
  __bf16* outfeat_bf = (__bf16*)take((size_t)(Tq - 1) * Bq * KOUT * 2);
  float*  h          = (float*) take((size_t)Bq * Hq * 4);
  __bf16* h_bf       = (__bf16*)take((size_t)Bq * Hq * 2);
  float*  hattn      = (float*) take((size_t)Bq * Hq * 4);
  float*  weighted   = (float*) take((size_t)Bq * TWOH * 4);
  __bf16* rnnin_bf   = (__bf16*)take((size_t)Bq * KIH * 2);
  float*  gi         = (float*) take((size_t)Bq * KIH * 4);
  float*  gh         = (float*) take((size_t)Bq * KIH * 4);

  // ---- one-time precompute ----
  hipMemcpyAsync(h, hidden, (size_t)Bq * Hq * 4, hipMemcpyDeviceToDevice, stream);

  {
    long n;
    n = (long)Vq * KOUT;  cvt_f32_bf16<<<gblocks(n), 256, 0, stream>>>(outW, outW_bf, n);
    n = (long)KIH * KIH;  cvt_f32_bf16<<<gblocks(n), 256, 0, stream>>>(Wih,  Wih_bf,  n);
    n = (long)KIH * Hq;   cvt_f32_bf16<<<gblocks(n), 256, 0, stream>>>(Whh,  Whh_bf,  n);
    n = (long)Hq * KIH;   cvt_f32_bf16<<<gblocks(n), 256, 0, stream>>>(attnW, attnW_bf, n);
    n = (long)Bq * Sq * TWOH;
    enc_transpose_bf16<<<gblocks(n), 256, 0, stream>>>(enc, encbt_bf);
  }

  // enc_proj[b,s,:] = enc_bt @ attn_W[:,H:3H]^T + attn_b   (time-invariant, bias folded)
  {
    dim3 grid(Hq / 128, (Bq * Sq) / 128);
    gemm_bf16_tn<<<grid, 256, 0, stream>>>(encbt_bf, TWOH,
                                           attnW_bf + Hq, KIH,
                                           encproj, Hq,
                                           Bq * Sq, Hq, TWOH, attnb);
  }

  // ---- sequential recurrence: 31 steps, cheap GEMMs + fused VALU ----
  for (int t = 0; t < Tq - 1; ++t) {
    __bf16* ofrow = outfeat_bf + (size_t)t * Bq * KOUT;

    gather_x<<<Bq, 128, 0, stream>>>(trg, embed, t, rnnin_bf, ofrow);

    long nh = (long)Bq * Hq;
    cvt_f32_bf16<<<gblocks(nh), 256, 0, stream>>>(h, h_bf, nh);

    // h_attn = h @ attn_W[:, :H]^T
    {
      dim3 grid(Hq / 128, 1);
      gemm_bf16_tn<<<grid, 256, 0, stream>>>(h_bf, Hq, attnW_bf, KIH,
                                             hattn, Hq, Bq, Hq, Hq, nullptr);
    }

    attn_fused<<<Bq, Sq, 0, stream>>>(hattn, encproj, vvec, src, enc,
                                      weighted, rnnin_bf, ofrow);

    // gi = [x|weighted] @ W_ih^T + b_ih
    {
      dim3 grid(KIH / 128, 1);
      gemm_bf16_tn<<<grid, 256, 0, stream>>>(rnnin_bf, KIH, Wih_bf, KIH,
                                             gi, KIH, Bq, KIH, KIH, bih);
    }
    // gh = h @ W_hh^T + b_hh
    {
      dim3 grid(KIH / 128, 1);
      gemm_bf16_tn<<<grid, 256, 0, stream>>>(h_bf, Hq, Whh_bf, Hq,
                                             gh, KIH, Bq, KIH, Hq, bhh);
    }

    gru_update<<<gblocks(nh), 256, 0, stream>>>(gi, gh, h, ofrow);
  }

  // ---- outputs[0] = 0 ; batched logits GEMM for t = 1..31 ----
  zero_f32<<<gblocks((long)Bq * Vq), 256, 0, stream>>>(out, (long)Bq * Vq);
  {
    const int M = (Tq - 1) * Bq;   // 1984 (multiple of 64)
    dim3 grid(Vq / 128, (M + 127) / 128);
    gemm_bf16_tn<<<grid, 256, 0, stream>>>(outfeat_bf, KOUT, outW_bf, KOUT,
                                           out + (size_t)Bq * Vq, Vq,
                                           M, Vq, KOUT, outb);
  }
}
